// Block_55972013801927
// MI455X (gfx1250) — matrix operand order repaired and checked by reading, not compiled, not run on hardware
//
#include <hip/hip_runtime.h>
#include <hip/hip_bf16.h>
#include <stdint.h>

// ---------------------------------------------------------------------------
// Transformer block (pre-LN attention + MLP) for MI455X / gfx1250.
// GEMMs: v_wmma_f32_16x16x32_f16 with async global->LDS staged B tiles
// (global_load_async_to_lds_b128 / s_wait_asynccnt), double buffered.
// ---------------------------------------------------------------------------

#define Bc 2
#define Tc 2048
#define Cc 1024
#define Hc 16
#define Dh 64
#define FFc 4096
#define BT (Bc * Tc)     // 4096

typedef __attribute__((ext_vector_type(16))) _Float16 v16h;
typedef __attribute__((ext_vector_type(8)))  _Float16 v8h;
typedef __attribute__((ext_vector_type(8)))  float    v8f;

#define NEG_INF (-__builtin_inff())

static __device__ __forceinline__ v8f wmma_f16(v16h a, v16h b, v8f c) {
    return __builtin_amdgcn_wmma_f32_16x16x32_f16(
        /*neg_a=*/false, a, /*neg_b=*/false, b,
        /*c_mod=*/(short)0, c, /*reuse_a=*/false, /*reuse_b=*/false);
}

// Raw LDS byte offset of a pointer into a __shared__ object (generic -> AS3).
typedef __attribute__((address_space(3))) void lds_void;
static __device__ __forceinline__ unsigned lds_addr(void* p) {
    return (unsigned)(uintptr_t)(lds_void*)p;
}

// Async DMA one b128 (16B) chunk from global into LDS (ASYNCcnt-tracked).
static __device__ __forceinline__ void async_copy_b128(unsigned lds_off,
                                                       const void* gptr) {
    unsigned long long ga = (unsigned long long)(uintptr_t)gptr;
    asm volatile("global_load_async_to_lds_b128 %0, %1, off"
                 :: "v"(lds_off), "v"(ga)
                 : "memory");
}
static __device__ __forceinline__ void wait_async_le(int n) {
    if (n == 0) asm volatile("s_wait_asynccnt 0x0" ::: "memory");
    else        asm volatile("s_wait_asynccnt 0x2" ::: "memory");
}

// Load A fragment (16x32 f16, row-major source, stride lda halves).
// Layout: lane<16 -> row=lane, K 0..15 ;
//         lane>=16 -> row=lane-16, K 16..31
static __device__ __forceinline__ v16h load_a_frag(const _Float16* base, size_t lda,
                                                   int lane) {
    const int lc = lane & 15, half = lane >> 4;
    const _Float16* ap = base + (size_t)lc * lda + half * 16;
    v16h a;
    ((v8h*)&a)[0] = *(const v8h*)(ap);
    ((v8h*)&a)[1] = *(const v8h*)(ap + 8);
    return a;
}

// Load B fragment (32x16 f16) from a row-major B^T (N x K) source, stride ldb.
// Layout: lane<16 -> col N=lane, K 0..15 ; lane>=16 -> col N=lane-16, K 16..31
static __device__ __forceinline__ v16h load_b_frag(const _Float16* baseT, size_t ldb,
                                                   int lane) {
    const int lc = lane & 15, half = lane >> 4;
    return *(const v16h*)(baseT + (size_t)lc * ldb + half * 16);
}

// ---------------------------------------------------------------------------
// Weight conversion kernels (f32 -> f16, pre-transposed for contiguous B frags)
// ---------------------------------------------------------------------------

// Wq/Wk/Wv: (H, C, D) f32  ->  Wt[(h*D+d)*C + c] f16   (i.e. B^T with K=C)
__global__ void cvt_qkv_kernel(const float* __restrict__ W, _Float16* __restrict__ Wt) {
    size_t idx = (size_t)blockIdx.x * 256 + threadIdx.x;   // H*C*D total
    int d = (int)(idx % Dh);
    size_t r = idx / Dh;
    int c = (int)(r % Cc);
    int h = (int)(r / Cc);
    Wt[((size_t)(h * Dh + d)) * Cc + c] = (_Float16)W[idx];
}

// Generic: in (R x N) f32 row-major -> out[n*R + r] f16 (B^T with K=R)
__global__ void cvt_tr_kernel(const float* __restrict__ W, _Float16* __restrict__ Wt,
                              int R, int N) {
    size_t idx = (size_t)blockIdx.x * 256 + threadIdx.x;   // R*N total
    int n = (int)(idx % N);
    size_t r = idx / N;
    Wt[(size_t)n * R + r] = (_Float16)W[idx];
}

// ---------------------------------------------------------------------------
// LayerNorm: one wave per row of C=1024; outputs f16 for WMMA consumption
// ---------------------------------------------------------------------------
__global__ __launch_bounds__(256) void ln_kernel(const float* __restrict__ x,
                                                 const float* __restrict__ g,
                                                 const float* __restrict__ beta,
                                                 _Float16* __restrict__ out) {
    int row  = blockIdx.x * 8 + (threadIdx.x >> 5);
    int lane = threadIdx.x & 31;
    const float* xr = x + (size_t)row * Cc;
    float s = 0.f, s2 = 0.f;
    #pragma unroll
    for (int i = lane; i < Cc; i += 32) { float v = xr[i]; s += v; s2 += v * v; }
    #pragma unroll
    for (int m = 16; m >= 1; m >>= 1) {
        s  += __shfl_xor(s,  m, 32);
        s2 += __shfl_xor(s2, m, 32);
    }
    float mu  = s * (1.0f / Cc);
    float inv = rsqrtf(s2 * (1.0f / Cc) - mu * mu + 1e-5f);
    _Float16* orow = out + (size_t)row * Cc;
    #pragma unroll
    for (int i = lane; i < Cc; i += 32)
        orow[i] = (_Float16)((xr[i] - mu) * inv * g[i] + beta[i]);
}

// ---------------------------------------------------------------------------
// WMMA GEMM: C[MxN] = A[MxK] (f16, row-major) * Bt[NxK]^T (f16) + epilogue
// Block = 128 threads (4 waves); block tile 128x64; wave tile 32x64.
// B tile (64 rows x 32 halves = 4KB) staged global->LDS with async DMA,
// double buffered so the copy of tile k+1 overlaps WMMA on tile k.
// EPI: 0 = store f16           (Q, K)
//      1 = store f16 scattered as V^T[b,h,d,t]  (V)
//      2 = +bias +resid -> f32 (attention output projection)
//      3 = +bias, ReLU -> f16  (FF1)
//      4 = +bias +resid -> f32 (FF2 -> final output)
// ---------------------------------------------------------------------------
template <int EPI>
__global__ __launch_bounds__(128) void gemm_wmma_kernel(
    const _Float16* __restrict__ A, const _Float16* __restrict__ Bt,
    const float* __restrict__ bias, const float* __restrict__ resid,
    _Float16* __restrict__ outH, float* __restrict__ outF,
    int M, int N, int K) {
    __shared__ _Float16 Bs[2][64 * 32];    // 2 x 4KB
    const int tid  = threadIdx.x;
    const int lane = tid & 31;
    const int wave = tid >> 5;
    const int lc   = lane & 15;
    const int half = lane >> 4;
    const int m0   = blockIdx.y * 128 + wave * 32;
    const int n0   = blockIdx.x * 64;

    // Async stage of B^T tile [n0..n0+63] x [k0..k0+31] into Bs[buf].
    // 128 threads x b128 covers 32 rows; each thread issues 2 copies.
    const int br = tid >> 2;            // 0..31
    const int bcol = (tid & 3) * 8;     // halves within row
    auto stage = [&](int buf, int k0) {
        const _Float16* g0 = Bt + (size_t)(n0 + br) * K + k0 + bcol;
        const _Float16* g1 = g0 + (size_t)32 * K;
        unsigned l0 = lds_addr(&Bs[buf][br * 32 + bcol]);
        async_copy_b128(l0, g0);
        async_copy_b128(l0 + 32 * 32 * 2, g1);
    };

    v8f acc0[4] = {};
    v8f acc1[4] = {};
    const _Float16* ap0 = A + (size_t)(m0 + lc) * K + half * 16;
    const _Float16* ap1 = ap0 + (size_t)16 * K;

    stage(0, 0);
    int cur = 0;
    for (int k0 = 0; k0 < K; k0 += 32) {
        const bool more = (k0 + 32) < K;
        if (more) stage(cur ^ 1, k0 + 32);
        wait_async_le(more ? 2 : 0);    // current tile landed in LDS
        __syncthreads();

        v16h a0, a1;
        ((v8h*)&a0)[0] = *(const v8h*)(ap0 + k0);
        ((v8h*)&a0)[1] = *(const v8h*)(ap0 + k0 + 8);
        ((v8h*)&a1)[0] = *(const v8h*)(ap1 + k0);
        ((v8h*)&a1)[1] = *(const v8h*)(ap1 + k0 + 8);
        #pragma unroll
        for (int t = 0; t < 4; ++t) {
            v16h bf = *(const v16h*)(&Bs[cur][(t * 16 + lc) * 32 + half * 16]);
            acc0[t] = wmma_f16(a0, bf, acc0[t]);
            acc1[t] = wmma_f16(a1, bf, acc1[t]);
        }
        __syncthreads();                // done reading Bs[cur] before restage
        cur ^= 1;
    }

    // C/D layout: vgpr r, lane<16 -> row +r, col lc ; lane>=16 -> row +r+8
    #pragma unroll
    for (int rt = 0; rt < 2; ++rt) {
        const v8f* accp = rt ? acc1 : acc0;
        const int mb = m0 + rt * 16;
        #pragma unroll
        for (int t = 0; t < 4; ++t) {
            #pragma unroll
            for (int r = 0; r < 8; ++r) {
                int m = mb + r + half * 8;
                int n = n0 + t * 16 + lc;
                float v = accp[t][r];
                if (EPI == 0) {
                    outH[(size_t)m * N + n] = (_Float16)v;
                } else if (EPI == 1) {
                    int bb = m >> 11, tt = m & (Tc - 1);   // m = b*T + t
                    int hh = n >> 6,  dd = n & (Dh - 1);   // n = h*D + d
                    outH[(((size_t)bb * Hc + hh) * Dh + dd) * Tc + tt] = (_Float16)v;
                } else if (EPI == 2) {
                    outF[(size_t)m * N + n] = v + bias[n] + resid[(size_t)m * N + n];
                } else if (EPI == 3) {
                    float y = v + bias[n];
                    outH[(size_t)m * N + n] = (_Float16)(y > 0.f ? y : 0.f);
                } else {
                    outF[(size_t)m * N + n] = v + bias[n] + resid[(size_t)m * N + n];
                }
            }
        }
    }
}

// ---------------------------------------------------------------------------
// Flash attention: one wave per (b, h, 16-row tile). Causal, scale = C^-0.5.
// q,k: f16 [B*T, C] (col = h*D+d, so per-head rows are contiguous along d)
// vt : f16 [B, H, D, T] (per-head V transposed -> contiguous B fragments)
// out: f16 [B*T, C]
// ---------------------------------------------------------------------------
__global__ __launch_bounds__(32) void attn_kernel(const _Float16* __restrict__ q,
                                                  const _Float16* __restrict__ k,
                                                  const _Float16* __restrict__ vt,
                                                  _Float16* __restrict__ attO) {
    __shared__ _Float16 P[16 * 32];
    const int lane = threadIdx.x & 31;
    const int lc   = lane & 15;
    const int half = lane >> 4;
    const int i0   = blockIdx.x * 16;
    const int h    = blockIdx.y;
    const int b    = blockIdx.z;

    const _Float16* Qh = q  + (size_t)b * Tc * Cc + (size_t)h * Dh;
    const _Float16* Kh = k  + (size_t)b * Tc * Cc + (size_t)h * Dh;
    const _Float16* Vh = vt + ((size_t)b * Hc + h) * Dh * Tc;

    float mrow[8], lrow[8];
    v8f o[4] = {};
    #pragma unroll
    for (int r = 0; r < 8; ++r) { mrow[r] = NEG_INF; lrow[r] = 0.f; }
    const float scale = 0.03125f;  // 1024^-0.5 == 1/32 exactly

    for (int s0 = 0; s0 < i0 + 16; s0 += 32) {
        // ---- scores: two 16x16 tiles (cols s0..s0+15, s0+16..s0+31) ----
        v8f sc[2] = {};
        #pragma unroll
        for (int st = 0; st < 2; ++st) {
            #pragma unroll
            for (int kd = 0; kd < Dh; kd += 32) {
                v16h af = load_a_frag(Qh + (size_t)i0 * Cc + kd, Cc, lane);
                v16h bf = load_b_frag(Kh + (size_t)(s0 + st * 16) * Cc + kd, Cc, lane);
                sc[st] = wmma_f16(af, bf, sc[st]);
            }
        }
        // ---- online softmax (row reductions across each 16-lane half) ----
        #pragma unroll
        for (int r = 0; r < 8; ++r) {
            int i  = i0 + r + half * 8;
            int j0 = s0 + lc;
            int j1 = s0 + 16 + lc;
            float e0 = (j0 <= i) ? sc[0][r] * scale : NEG_INF;
            float e1 = (j1 <= i) ? sc[1][r] * scale : NEG_INF;
            float mx = fmaxf(e0, e1);
            #pragma unroll
            for (int msk = 8; msk >= 1; msk >>= 1)
                mx = fmaxf(mx, __shfl_xor(mx, msk, 32));
            float mnew = fmaxf(mrow[r], mx);
            float p0 = __expf(e0 - mnew);
            float p1 = __expf(e1 - mnew);
            float rs = p0 + p1;
            #pragma unroll
            for (int msk = 8; msk >= 1; msk >>= 1)
                rs += __shfl_xor(rs, msk, 32);
            float corr = __expf(mrow[r] - mnew);
            lrow[r] = lrow[r] * corr + rs;
            mrow[r] = mnew;
            #pragma unroll
            for (int t = 0; t < 4; ++t) o[t][r] *= corr;
            P[(r + half * 8) * 32 + lc]      = (_Float16)p0;
            P[(r + half * 8) * 32 + 16 + lc] = (_Float16)p1;
        }
        __syncthreads();
        // ---- att x V: probs re-read from LDS as A fragment, 4 d-tiles ----
        v16h pf;
        const _Float16* pp = P + lc * 32 + half * 16;
        ((v8h*)&pf)[0] = *(const v8h*)(pp);
        ((v8h*)&pf)[1] = *(const v8h*)(pp + 8);
        #pragma unroll
        for (int t = 0; t < 4; ++t) {
            v16h bf = *(const v16h*)(Vh + (size_t)(t * 16 + lc) * Tc + s0 + half * 16);
            o[t] = wmma_f16(pf, bf, o[t]);
        }
        __syncthreads();
    }
    // ---- normalize + store ----
    #pragma unroll
    for (int t = 0; t < 4; ++t) {
        #pragma unroll
        for (int r = 0; r < 8; ++r) {
            int i = i0 + r + half * 8;
            int n = t * 16 + lc;
            attO[((size_t)b * Tc + i) * Cc + (size_t)h * Dh + n] =
                (_Float16)(o[t][r] / lrow[r]);
        }
    }
}

// ---------------------------------------------------------------------------
// Host-side orchestration
// ---------------------------------------------------------------------------
extern "C" void kernel_launch(void* const* d_in, const int* in_sizes, int n_in,
                              void* d_out, int out_size, void* d_ws, size_t ws_size,
                              hipStream_t stream) {
    const float* x   = (const float*)d_in[0];
    const float* Wq  = (const float*)d_in[1];
    const float* Wk  = (const float*)d_in[2];
    const float* Wv  = (const float*)d_in[3];
    const float* Wo  = (const float*)d_in[4];
    const float* bo  = (const float*)d_in[5];
    const float* W1  = (const float*)d_in[6];
    const float* b1  = (const float*)d_in[7];
    const float* W2  = (const float*)d_in[8];
    const float* b2  = (const float*)d_in[9];
    const float* g1  = (const float*)d_in[10];
    const float* be1 = (const float*)d_in[11];
    const float* g2  = (const float*)d_in[12];
    const float* be2 = (const float*)d_in[13];
    float* out = (float*)d_out;

    // ---- workspace carving (~120 MB) ----
    uint8_t* ws = (uint8_t*)d_ws;
    size_t off = 0;
    auto carve = [&](size_t bytes) -> void* {
        void* p = ws + off;
        off = (off + bytes + 255) & ~(size_t)255;
        return p;
    };
    _Float16* h16  = (_Float16*)carve((size_t)BT * Cc * 2);
    _Float16* q16  = (_Float16*)carve((size_t)BT * Cc * 2);
    _Float16* k16  = (_Float16*)carve((size_t)BT * Cc * 2);
    _Float16* vt16 = (_Float16*)carve((size_t)BT * Cc * 2);
    _Float16* aO16 = (_Float16*)carve((size_t)BT * Cc * 2);
    _Float16* h216 = (_Float16*)carve((size_t)BT * Cc * 2);
    _Float16* a116 = (_Float16*)carve((size_t)BT * FFc * 2);
    _Float16* wqt  = (_Float16*)carve((size_t)Cc * Cc * 2);
    _Float16* wkt  = (_Float16*)carve((size_t)Cc * Cc * 2);
    _Float16* wvt  = (_Float16*)carve((size_t)Cc * Cc * 2);
    _Float16* wot  = (_Float16*)carve((size_t)Cc * Cc * 2);
    _Float16* w1t  = (_Float16*)carve((size_t)Cc * FFc * 2);
    _Float16* w2t  = (_Float16*)carve((size_t)FFc * Cc * 2);
    float*    x2   = (float*)carve((size_t)BT * Cc * 4);

    // ---- weight packing (f32 -> f16, transposed) ----
    cvt_qkv_kernel<<<(Hc * Cc * Dh) / 256, 256, 0, stream>>>(Wq, wqt);
    cvt_qkv_kernel<<<(Hc * Cc * Dh) / 256, 256, 0, stream>>>(Wk, wkt);
    cvt_qkv_kernel<<<(Hc * Cc * Dh) / 256, 256, 0, stream>>>(Wv, wvt);
    cvt_tr_kernel<<<(Cc * Cc) / 256, 256, 0, stream>>>(Wo, wot, Cc, Cc);
    cvt_tr_kernel<<<(Cc * FFc) / 256, 256, 0, stream>>>(W1, w1t, Cc, FFc);
    cvt_tr_kernel<<<(FFc * Cc) / 256, 256, 0, stream>>>(W2, w2t, FFc, Cc);

    // ---- LN1 ----
    ln_kernel<<<BT / 8, 256, 0, stream>>>(x, g1, be1, h16);

    // ---- QKV projections ----
    dim3 blk(128);
    dim3 grdC(Cc / 64, BT / 128);   // N=1024, block tile 128x64
    gemm_wmma_kernel<0><<<grdC, blk, 0, stream>>>(h16, wqt, nullptr, nullptr,
                                                  q16, nullptr, BT, Cc, Cc);
    gemm_wmma_kernel<0><<<grdC, blk, 0, stream>>>(h16, wkt, nullptr, nullptr,
                                                  k16, nullptr, BT, Cc, Cc);
    gemm_wmma_kernel<1><<<grdC, blk, 0, stream>>>(h16, wvt, nullptr, nullptr,
                                                  vt16, nullptr, BT, Cc, Cc);

    // ---- attention (flash, causal) ----
    attn_kernel<<<dim3(Tc / 16, Hc, Bc), 32, 0, stream>>>(q16, k16, vt16, aO16);

    // ---- output projection + bias + residual -> x2 (f32) ----
    gemm_wmma_kernel<2><<<grdC, blk, 0, stream>>>(aO16, wot, bo, x,
                                                  nullptr, x2, BT, Cc, Cc);

    // ---- LN2 ----
    ln_kernel<<<BT / 8, 256, 0, stream>>>(x2, g2, be2, h216);

    // ---- FFN ----
    dim3 grdF(FFc / 64, BT / 128);  // N=4096
    gemm_wmma_kernel<3><<<grdF, blk, 0, stream>>>(h216, w1t, b1, nullptr,
                                                  a116, nullptr, BT, FFc, Cc);
    gemm_wmma_kernel<4><<<grdC, blk, 0, stream>>>(a116, w2t, b2, x2,
                                                  nullptr, out, BT, Cc, FFc);

    (void)in_sizes; (void)n_in; (void)out_size; (void)ws_size;
}
